// MultiHeadAttentionBlock_61521111547975
// MI455X (gfx1250) — compile-verified, matrix operand order checked by reading, not run on hardware
//
#include <hip/hip_runtime.h>

// Problem constants (match reference)
#define B_  2
#define L_  2048
#define D_  1024
#define H_  16
#define DK_ 64
#define M_  (B_ * L_)   // 4096 rows total

typedef __attribute__((ext_vector_type(16))) __bf16 v16bf;
typedef __attribute__((ext_vector_type(8)))  __bf16 v8bf;
typedef __attribute__((ext_vector_type(8)))  float  v8f;
typedef __attribute__((ext_vector_type(4)))  float  v4f;

// ---------------------------------------------------------------------------
// bf16 fragment loader for V_WMMA_F32_16X16X32_BF16.
// 16-bit A/B layout (ISA 7.12.2): lane&15 selects the row (A) / column (B),
// lane>>4 selects which K-half; halves 0..7 = K[h*8 .. h*8+7],
// halves 8..15 = K[h*8+16 .. h*8+23].  Two 16B global/ds loads per lane.
// ---------------------------------------------------------------------------
__device__ inline v16bf load_ab_bf16(const __bf16* base, int ld) {
  int lane = threadIdx.x & 31;
  const __bf16* p = base + (size_t)(lane & 15) * ld + (lane >> 4) * 8;
  v8bf lo = *(const v8bf*)(p);
  v8bf hi = *(const v8bf*)(p + 16);
  v16bf f;
#pragma unroll
  for (int i = 0; i < 8; ++i) { f[i] = lo[i]; f[8 + i] = hi[i]; }
  return f;
}

#define WMMA_BF16(A, Bm, C) \
  __builtin_amdgcn_wmma_f32_16x16x32_bf16(false, (A), false, (Bm), (short)0, (C), false, false)

// CDNA5 async global->LDS copy (ASYNCcnt-tracked), 16B per lane.
__device__ inline void async_b128_to_lds(const __bf16* gsrc, __bf16* lds_dst) {
  unsigned ldsa = (unsigned)(size_t)lds_dst;   // generic addr low 32b = LDS offset
  asm volatile("global_load_async_to_lds_b128 %0, %1, off"
               :: "v"(ldsa), "v"(gsrc) : "memory");
}
__device__ inline void wait_asynccnt0() {
  asm volatile("s_wait_asynccnt 0x0" ::: "memory");
}

// ---------------------------------------------------------------------------
// One-shot f32 -> bf16 conversion (vectorized: v8f load, 4x v_cvt_pk, b128 st)
// ---------------------------------------------------------------------------
__global__ __launch_bounds__(256) void cvt_f32_bf16_kernel(
    const float* __restrict__ X, __bf16* __restrict__ Y, int n) {
  int i = (blockIdx.x * 256 + threadIdx.x) * 8;
  if (i >= n) return;
  v8f x = *(const v8f*)(X + i);
  v8bf y;
#pragma unroll
  for (int j = 0; j < 8; ++j) y[j] = (__bf16)x[j];
  *(v8bf*)(Y + i) = y;
}

// ---------------------------------------------------------------------------
// Projection kernel: Y = X @ W^T  (M=4096, N=D=1024, K=D=1024), pure bf16.
// One wave computes a 32x64 tile: 2 A frags x 4 B frags = 8 WMMAs per K-chunk
// (8 independent accumulator chains -> no WMMA RAW nops).  Epilogue:
//   mode 0/1 (Q/K): fused interleaved RoPE, store bf16 [B,H,L,DK]
//   mode 2   (V)  : store transposed bf16 [B,H,DK,L]
// ---------------------------------------------------------------------------
__global__ __launch_bounds__(256) void proj_rope_kernel(
    const __bf16* __restrict__ X, const __bf16* __restrict__ W,
    __bf16* __restrict__ Y, int mode) {
  const int wid  = threadIdx.x >> 5;
  const int lane = threadIdx.x & 31;
  const int tile = blockIdx.x * 8 + wid;         // (M_/32) * (D_/64) tiles
  const int ntn  = D_ / 64;
  const int m0   = (tile / ntn) * 32;
  const int n0   = (tile % ntn) * 64;

  v8f acc[2][4] = {};
  for (int kk = 0; kk < D_; kk += 32) {
    if (kk + 32 < D_) {
      __builtin_prefetch(X + (size_t)m0 * D_ + kk + 32, 0, 3);
      __builtin_prefetch(W + (size_t)n0 * D_ + kk + 32, 0, 3);
    }
    v16bf a0 = load_ab_bf16(X + (size_t)m0 * D_ + kk, D_);
    v16bf a1 = load_ab_bf16(X + (size_t)(m0 + 16) * D_ + kk, D_);
#pragma unroll
    for (int t = 0; t < 4; ++t) {
      v16bf bm = load_ab_bf16(W + (size_t)(n0 + t * 16) * D_ + kk, D_);
      acc[0][t] = WMMA_BF16(a0, bm, acc[0][t]);
      acc[1][t] = WMMA_BF16(a1, bm, acc[1][t]);
    }
  }

  // C layout: element (row = j + 8*(lane>>4), col = lane&15) in acc[r][t][j]
  const int cl = lane & 15, hl = lane >> 4;
#pragma unroll
  for (int r = 0; r < 2; ++r) {
#pragma unroll
    for (int t = 0; t < 4; ++t) {
      const int col = n0 + t * 16 + cl;          // feature in [0, D)
      const int h   = col >> 6;
      const int dk  = col & (DK_ - 1);
#pragma unroll
      for (int j = 0; j < 8; ++j) {
        const int row = m0 + r * 16 + j + 8 * hl;
        const int bi  = row >> 11;               // / L_
        const int pos = row & (L_ - 1);
        float x = acc[r][t][j];
        float y;
        if (mode != 2) {
          // interleaved RoPE: pair (2i, 2i+1) sits on adjacent lanes
          float xp = __shfl_xor(x, 1);
          int   i2 = dk & ~1;                    // 2*i
          float ang = (float)pos * __expf(-(float)i2 * (9.210340371976184f / (float)DK_));
          float c = __cosf(ang), s = __sinf(ang);
          y = (dk & 1) ? fmaf(x, c, xp * s) : fmaf(x, c, -xp * s);
        } else {
          y = x;
        }
        size_t addr;
        if (mode == 2)  // V transposed: [B,H,DK,L]
          addr = ((size_t)(bi * H_ + h) * DK_ + dk) * L_ + pos;
        else            // Q/K: [B,H,L,DK]
          addr = ((size_t)(bi * H_ + h) * L_ + pos) * DK_ + dk;
        Y[addr] = (__bf16)y;
      }
    }
  }
}

// ---------------------------------------------------------------------------
// Flash attention.  A block owns 128 consecutive queries of one (b,h): 8
// waves x 16-query tiles.  Each 32-key chunk of K (32x64) and V^T (64x32) is
// staged into LDS once per block with CDNA5 async global->LDS b128 copies
// (ASYNCcnt), then all 8 waves build WMMA fragments from LDS: 8x reuse.
//   S = Q K^T (4 WMMAs) -> online softmax (16-lane shuffle reductions)
//   P relayout via per-wave LDS scratch (DS ops are in-order per wave)
//   O += P V (4 WMMAs).  Output bf16 [B, L, D].
// ---------------------------------------------------------------------------
__global__ __launch_bounds__(256) void flash_attn_kernel(
    const __bf16* __restrict__ Q, const __bf16* __restrict__ K,
    const __bf16* __restrict__ Vt, const int* __restrict__ mask,
    __bf16* __restrict__ O) {
  __shared__ __attribute__((aligned(16))) __bf16 kt[32 * 64];       // [key][feat]
  __shared__ __attribute__((aligned(16))) __bf16 vt[64 * 32];       // [feat][key]
  __shared__ __attribute__((aligned(16))) __bf16 plds[8][16 * 32];  // per-wave P

  const int tid  = threadIdx.x;
  const int wid  = tid >> 5;
  const int lane = tid & 31;
  const int nqb  = L_ / 128;                     // query-blocks per (b,h)
  const int bh   = blockIdx.x / nqb;             // 0 .. B_*H_-1
  const int q0   = (blockIdx.x % nqb) * 128 + wid * 16;

  const __bf16* Qb = Q  + ((size_t)bh * L_ + q0) * DK_;
  const __bf16* Kb = K  + (size_t)bh * L_ * DK_;
  const __bf16* Vb = Vt + (size_t)bh * DK_ * L_;

  // Per-thread cooperative-copy coordinates (16B per thread per tile)
  const int krow = tid >> 3, kcol = (tid & 7) * 8;   // K: 32 rows x 64 feats
  const int vrow = tid >> 2, vcol = (tid & 3) * 8;   // V: 64 feats x 32 keys
  const __bf16* kg = Kb + (size_t)krow * DK_ + kcol;
  const __bf16* vg = Vb + (size_t)vrow * L_ + vcol;
  __bf16* klds = kt + krow * 64 + kcol;
  __bf16* vlds = vt + vrow * 32 + vcol;

  const v16bf qa0 = load_ab_bf16(Qb + 0,  DK_);
  const v16bf qa1 = load_ab_bf16(Qb + 32, DK_);

  v8f   o[4] = {};
  float rm[8], rl[8];
#pragma unroll
  for (int j = 0; j < 8; ++j) { rm[j] = -3.0e38f; rl[j] = 0.0f; }

  const int   cl = lane & 15, hl = lane >> 4;
  const float scale = 0.125f;                    // 1/sqrt(DK)

  for (int k0 = 0; k0 < L_; k0 += 32) {
    // ---- stage K/V chunk into LDS (async, ASYNCcnt-tracked) ----
    __syncthreads();                             // everyone done with prev tiles
    async_b128_to_lds(kg + (size_t)k0 * DK_, klds);
    async_b128_to_lds(vg + k0, vlds);
    wait_asynccnt0();
    __syncthreads();                             // tiles visible to all waves

    // ---- S = Q K^T for 16 queries x 32 keys (two 16x16 C fragments) ----
    v8f s[2] = {};
#pragma unroll
    for (int ng = 0; ng < 2; ++ng) {
      v16bf kb0 = load_ab_bf16(kt + (ng * 16) * 64 + 0,  64);
      v16bf kb1 = load_ab_bf16(kt + (ng * 16) * 64 + 32, 64);
      s[ng] = WMMA_BF16(qa0, kb0, s[ng]);
      s[ng] = WMMA_BF16(qa1, kb1, s[ng]);
    }
    // ---- scale + mask ----
#pragma unroll
    for (int ng = 0; ng < 2; ++ng) {
      const int kc = k0 + ng * 16 + cl;
#pragma unroll
      for (int j = 0; j < 8; ++j) {
        const int qr = q0 + j + 8 * hl;
        float sv = s[ng][j] * scale;
        if (mask[(size_t)qr * L_ + kc] == 0) sv = -3.0e38f;
        s[ng][j] = sv;
      }
    }
    // ---- online softmax (row stats via 16-lane xor reductions) ----
    float corr[8];
#pragma unroll
    for (int j = 0; j < 8; ++j) {
      float mnew = fmaxf(s[0][j], s[1][j]);
#pragma unroll
      for (int mm = 1; mm < 16; mm <<= 1) mnew = fmaxf(mnew, __shfl_xor(mnew, mm));
      float mtot = fmaxf(rm[j], mnew);
      corr[j] = __expf(rm[j] - mtot);
      rm[j] = mtot;
      float p0 = __expf(s[0][j] - mtot);
      float p1 = __expf(s[1][j] - mtot);
      s[0][j] = p0; s[1][j] = p1;
      float rs = p0 + p1;
#pragma unroll
      for (int mm = 1; mm < 16; mm <<= 1) rs += __shfl_xor(rs, mm);
      rl[j] = rl[j] * corr[j] + rs;
    }
#pragma unroll
    for (int t = 0; t < 4; ++t)
#pragma unroll
      for (int j = 0; j < 8; ++j) o[t][j] *= corr[j];

    // ---- P: C-fragment -> per-wave LDS (16x32) -> A fragment.  DS ops are
    //      in-order within a wave; plds[wid] is wave-private (no barrier). ----
#pragma unroll
    for (int ng = 0; ng < 2; ++ng)
#pragma unroll
      for (int j = 0; j < 8; ++j)
        plds[wid][(j + 8 * hl) * 32 + ng * 16 + cl] = (__bf16)s[ng][j];
    const v16bf pa = load_ab_bf16(&plds[wid][0], 32);

    // ---- O += P V from the shared V^T tile ----
#pragma unroll
    for (int t = 0; t < 4; ++t) {
      v16bf vb = load_ab_bf16(vt + (t * 16) * 32, 32);
      o[t] = WMMA_BF16(pa, vb, o[t]);
    }
  }

  // ---- normalize and store bf16 to [B, L, D] ----
  const int h  = bh & (H_ - 1);
  const int bb = bh >> 4;
#pragma unroll
  for (int t = 0; t < 4; ++t)
#pragma unroll
    for (int j = 0; j < 8; ++j) {
      const int qr = q0 + j + 8 * hl;
      float val = o[t][j] / rl[j];
      O[((size_t)bb * L_ + qr) * D_ + h * DK_ + t * 16 + cl] = (__bf16)val;
    }
}

// ---------------------------------------------------------------------------
// Final projection: out = AO @ w_o^T, AO bf16, w_o bf16 (pre-converted),
// out f32 [B, L, D].  Same 32x64-per-wave tiling.
// ---------------------------------------------------------------------------
__global__ __launch_bounds__(256) void out_proj_kernel(
    const __bf16* __restrict__ X, const __bf16* __restrict__ W,
    float* __restrict__ Y) {
  const int wid  = threadIdx.x >> 5;
  const int lane = threadIdx.x & 31;
  const int tile = blockIdx.x * 8 + wid;
  const int ntn  = D_ / 64;
  const int m0   = (tile / ntn) * 32;
  const int n0   = (tile % ntn) * 64;

  v8f acc[2][4] = {};
  for (int kk = 0; kk < D_; kk += 32) {
    if (kk + 32 < D_) {
      __builtin_prefetch(X + (size_t)m0 * D_ + kk + 32, 0, 3);
      __builtin_prefetch(W + (size_t)n0 * D_ + kk + 32, 0, 3);
    }
    v16bf a0 = load_ab_bf16(X + (size_t)m0 * D_ + kk, D_);
    v16bf a1 = load_ab_bf16(X + (size_t)(m0 + 16) * D_ + kk, D_);
#pragma unroll
    for (int t = 0; t < 4; ++t) {
      v16bf bm = load_ab_bf16(W + (size_t)(n0 + t * 16) * D_ + kk, D_);
      acc[0][t] = WMMA_BF16(a0, bm, acc[0][t]);
      acc[1][t] = WMMA_BF16(a1, bm, acc[1][t]);
    }
  }
  const int cl = lane & 15, hl = lane >> 4;
#pragma unroll
  for (int r = 0; r < 2; ++r)
#pragma unroll
    for (int t = 0; t < 4; ++t)
#pragma unroll
      for (int j = 0; j < 8; ++j)
        Y[(size_t)(m0 + r * 16 + j + 8 * hl) * D_ + n0 + t * 16 + cl] = acc[r][t][j];
}

// ---------------------------------------------------------------------------
extern "C" void kernel_launch(void* const* d_in, const int* in_sizes, int n_in,
                              void* d_out, int out_size, void* d_ws, size_t ws_size,
                              hipStream_t stream) {
  const float* q    = (const float*)d_in[0];
  const float* k    = (const float*)d_in[1];
  const float* v    = (const float*)d_in[2];
  const int*   mask = (const int*)d_in[3];
  const float* w_q  = (const float*)d_in[4];
  const float* w_k  = (const float*)d_in[5];
  const float* w_v  = (const float*)d_in[6];
  const float* w_o  = (const float*)d_in[7];
  float* out = (float*)d_out;

  const size_t MD = (size_t)M_ * D_;   // 4M elems
  const size_t DD = (size_t)D_ * D_;   // 1M elems

  // Workspace layout (all bf16):
  // Qr | Kr | Vt | AO (4M each) | Xq | Xk | Xv (4M each) | Wq | Wk | Wv | Wo (1M each)
  __bf16* Qr = (__bf16*)d_ws;
  __bf16* Kr = Qr + MD;
  __bf16* Vt = Kr + MD;
  __bf16* AO = Vt + MD;
  __bf16* Xq = AO + MD;
  __bf16* Xk = Xq + MD;
  __bf16* Xv = Xk + MD;
  __bf16* Wq = Xv + MD;
  __bf16* Wk = Wq + DD;
  __bf16* Wv = Wk + DD;
  __bf16* Wo = Wv + DD;

  dim3 blk(256);

  // One-shot f32 -> bf16 conversions (inputs + weights)
  dim3 grd_cx((unsigned)(MD / 8 / 256));
  dim3 grd_cw((unsigned)(DD / 8 / 256));
  cvt_f32_bf16_kernel<<<grd_cx, blk, 0, stream>>>(q,   Xq, (int)MD);
  cvt_f32_bf16_kernel<<<grd_cx, blk, 0, stream>>>(k,   Xk, (int)MD);
  cvt_f32_bf16_kernel<<<grd_cx, blk, 0, stream>>>(v,   Xv, (int)MD);
  cvt_f32_bf16_kernel<<<grd_cw, blk, 0, stream>>>(w_q, Wq, (int)DD);
  cvt_f32_bf16_kernel<<<grd_cw, blk, 0, stream>>>(w_k, Wk, (int)DD);
  cvt_f32_bf16_kernel<<<grd_cw, blk, 0, stream>>>(w_v, Wv, (int)DD);
  cvt_f32_bf16_kernel<<<grd_cw, blk, 0, stream>>>(w_o, Wo, (int)DD);

  const int proj_tiles = (M_ / 32) * (D_ / 64);  // 2048 waves
  dim3 grd_proj(proj_tiles / 8);                 // 256 blocks, 8 waves each
  dim3 grd_attn(B_ * H_ * (L_ / 128));           // 512 blocks, 8 waves each

  proj_rope_kernel<<<grd_proj, blk, 0, stream>>>(Xq, Wq, Qr, 0);
  proj_rope_kernel<<<grd_proj, blk, 0, stream>>>(Xk, Wk, Kr, 1);
  proj_rope_kernel<<<grd_proj, blk, 0, stream>>>(Xv, Wv, Vt, 2);
  flash_attn_kernel<<<grd_attn, blk, 0, stream>>>(Qr, Kr, Vt, mask, AO);
  out_proj_kernel<<<grd_proj, blk, 0, stream>>>(AO, Wo, out);
}
